// SmallWorldLayer_6330781794646
// MI455X (gfx1250) — matrix operand order repaired and checked by reading, not compiled, not run on hardware
//
#include <hip/hip_runtime.h>
#include <hip/hip_bf16.h>

typedef __attribute__((ext_vector_type(16))) __bf16 v16bf;
typedef __attribute__((ext_vector_type(8)))  float  v8f;
typedef __attribute__((ext_vector_type(4)))  int    v4i;

#define IN_DIM   256
#define OUT_DIM  256
#define NCONN    6553
#define M_ROWS   (16 * 16384)      // 262144
#define M_TILES  (M_ROWS / 16)     // 16384
#define GEMM_BLOCKS 2048

#if defined(__AMDGCN__) && __has_builtin(__builtin_amdgcn_global_load_async_to_lds_b128) && __has_builtin(__builtin_amdgcn_s_wait_asynccnt)
#define USE_ASYNC_LDS 1
// Builtin signature (from hipcc diagnostic): (v4i AS1*, v4i AS3*, imm, imm)
typedef __attribute__((address_space(1))) v4i* gas_p;
typedef __attribute__((address_space(3))) v4i* las_p;
#define G_AS1(p) ((gas_p)(__UINTPTR_TYPE__)(p))
#define L_AS3(p) ((las_p)(unsigned int)(__UINTPTR_TYPE__)(p))
#else
#define USE_ASYNC_LDS 0
#endif

// ---------------------------------------------------------------------------
// B-fragment swizzle: map weight element (o = N index, k = K index) to its
// slot in the pre-swizzled bf16 fragment buffer. Layout per CDNA5 ISA
// 16-bit B-matrix (32x16) for v_wmma_f32_16x16x32_bf16:
//   lane = ((k_local/8)%2)*16 + (o%16);  VGPR v packs K = 16*(v/4)+8*(lane/16)+2*(v%4),+1
// Fragments stored as [ntile][kchunk][lane][16 bf16] (32 bytes/lane).
// ---------------------------------------------------------------------------
__device__ __forceinline__ int frag_index(int o, int k) {
    int nt   = o >> 4;
    int lm   = o & 15;
    int kc   = k >> 5;
    int rem  = k & 31;
    int half = (rem >> 3) & 1;
    int r2   = (rem & 7) | ((rem >> 4) << 3);        // v*2 + p
    int lane = half * 16 + lm;
    return ((nt * 8 + kc) * 32 + lane) * 16 + r2;
}

__global__ void swl_build_bfrags(const float* __restrict__ W,
                                 __bf16* __restrict__ frags) {
    int t = blockIdx.x * blockDim.x + threadIdx.x;   // 65536 threads
    int o = t >> 8;
    int k = t & 255;
    frags[frag_index(o, k)] = (__bf16)W[t];
}

// Small-world rewiring scatter ("set" semantics folded in):
// W_eff[r][c] = W[r][c] + 0.1*value.
__global__ void swl_scatter(const float* __restrict__ W,
                            const int* __restrict__ rows,
                            const int* __restrict__ cols,
                            const float* __restrict__ vals,
                            __bf16* __restrict__ frags) {
    int c = blockIdx.x * blockDim.x + threadIdx.x;
    if (c >= NCONN) return;
    int o = rows[c];
    int k = cols[c];
    float v = W[o * IN_DIM + k] + 0.1f * vals[c];
    frags[frag_index(o, k)] = (__bf16)v;
}

// Issue one 16KB x tile (16 rows x 256 f32) into an LDS buffer.
// Each of 256 threads moves 64 contiguous bytes (4 x b128).
__device__ __forceinline__ void tile_issue(const float* __restrict__ xb,
                                           float* __restrict__ dst, int tid) {
#if USE_ASYNC_LDS
    const float* g = xb + tid * 16;
    float*       l = dst + tid * 16;
    __builtin_amdgcn_global_load_async_to_lds_b128(G_AS1(g),      L_AS3(l),      0, 0);
    __builtin_amdgcn_global_load_async_to_lds_b128(G_AS1(g + 4),  L_AS3(l + 4),  0, 0);
    __builtin_amdgcn_global_load_async_to_lds_b128(G_AS1(g + 8),  L_AS3(l + 8),  0, 0);
    __builtin_amdgcn_global_load_async_to_lds_b128(G_AS1(g + 12), L_AS3(l + 12), 0, 0);
#else
#pragma unroll
    for (int i = 0; i < 4; ++i) {
        int idx = tid + i * 256;             // 1024 float4 = 16KB
        ((float4*)dst)[idx] = ((const float4*)xb)[idx];
    }
#endif
}

// ---------------------------------------------------------------------------
// GEMM: out[m][n] = sum_k x[m][k] * W_eff[n][k] + b[n]
// Block = 256 threads = 8 waves. Wave w owns N in [32w, 32w+32) (2 N-tiles),
// holds all 16 B fragments (2 ntiles x 8 kchunks) register-resident.
// Block grid-strides over 16-row M tiles; x tile double-buffered in LDS via
// async load-to-LDS (ASYNCcnt).
// ---------------------------------------------------------------------------
__global__ __launch_bounds__(256)
void swl_gemm(const float* __restrict__ x,
              const float* __restrict__ bias,
              const __bf16* __restrict__ frags,
              float* __restrict__ out) {
    __shared__ float xs[2][16 * IN_DIM];         // 2 x 16 KB tiles

    const int tid  = threadIdx.x;
    const int w    = tid >> 5;                   // wave id 0..7
    const int lane = tid & 31;
    const int lm   = lane & 15;
    const int half = lane >> 4;

    // Register-resident B fragments (~128 VGPRs).
    v16bf Bf[2][8];
#pragma unroll
    for (int j = 0; j < 2; ++j) {
        const int nt = 2 * w + j;
#pragma unroll
        for (int kc = 0; kc < 8; ++kc) {
            const v16bf* p =
                (const v16bf*)(frags + ((nt * 8 + kc) * 32 + lane) * 16);
            Bf[j][kc] = *p;
        }
    }

    const float bias0 = bias[32 * w + lm];
    const float bias1 = bias[32 * w + 16 + lm];

    int mt  = blockIdx.x;
    int buf = 0;
    if (mt < M_TILES)
        tile_issue(x + (size_t)mt * 16 * IN_DIM, xs[0], tid);

    for (; mt < M_TILES; mt += gridDim.x) {
        const int  nmt       = mt + (int)gridDim.x;
        const bool have_next = (nmt < M_TILES);
        if (have_next)
            tile_issue(x + (size_t)nmt * 16 * IN_DIM, xs[buf ^ 1], tid);

#if USE_ASYNC_LDS
        // Async loads retire in order: <=4 outstanding means the current
        // tile's 4 loads are done; the next tile's 4 may still be in flight.
        if (have_next) __builtin_amdgcn_s_wait_asynccnt(4);
        else           __builtin_amdgcn_s_wait_asynccnt(0);
#endif
        __syncthreads();                         // tile visible to all waves

        v8f acc0 = {};
        v8f acc1 = {};

        // A-fragment (16-bit A 16x32 layout): lane holds row M = lane%16,
        // K = kc*32 + g*16 + half*8 + (0..7) for VGPR groups g = 0,1.
        const float* rowp = &xs[buf][lm * IN_DIM + half * 8];
#pragma unroll
        for (int kc = 0; kc < 8; ++kc) {
            v16bf A;
#pragma unroll
            for (int g = 0; g < 2; ++g) {
                const float4 f0 = *(const float4*)(rowp + kc * 32 + g * 16);
                const float4 f1 = *(const float4*)(rowp + kc * 32 + g * 16 + 4);
                A[g * 8 + 0] = (__bf16)f0.x;
                A[g * 8 + 1] = (__bf16)f0.y;
                A[g * 8 + 2] = (__bf16)f0.z;
                A[g * 8 + 3] = (__bf16)f0.w;
                A[g * 8 + 4] = (__bf16)f1.x;
                A[g * 8 + 5] = (__bf16)f1.y;
                A[g * 8 + 6] = (__bf16)f1.z;
                A[g * 8 + 7] = (__bf16)f1.w;
            }
            acc0 = __builtin_amdgcn_wmma_f32_16x16x32_bf16(
                false, A, false, Bf[0][kc], (short)0, acc0, false, false);
            acc1 = __builtin_amdgcn_wmma_f32_16x16x32_bf16(
                false, A, false, Bf[1][kc], (short)0, acc1, false, false);
        }

        // Epilogue. C/D 16x16 f32 layout: VGPR j -> M = j + 8*(lane/16),
        // N = lane%16 within the tile.
        float* outp = out + (size_t)mt * 16 * OUT_DIM + 32 * w + lm;
#pragma unroll
        for (int j = 0; j < 8; ++j) {
            const size_t row = (size_t)(j + 8 * half) * OUT_DIM;
            outp[row]      = acc0[j] + bias0;
            outp[row + 16] = acc1[j] + bias1;
        }

        __syncthreads();   // compute done before this buffer is re-filled
        buf ^= 1;
    }
}

extern "C" void kernel_launch(void* const* d_in, const int* in_sizes, int n_in,
                              void* d_out, int out_size, void* d_ws, size_t ws_size,
                              hipStream_t stream) {
    const float* x    = (const float*)d_in[0];
    const float* W    = (const float*)d_in[1];
    const float* b    = (const float*)d_in[2];
    const int*   rows = (const int*)d_in[3];
    const int*   cols = (const int*)d_in[4];
    const float* vals = (const float*)d_in[5];
    float*       out  = (float*)d_out;

    __bf16* frags = (__bf16*)d_ws;   // 65536 bf16 = 128 KB

    swl_build_bfrags<<<(OUT_DIM * IN_DIM) / 256, 256, 0, stream>>>(W, frags);
    swl_scatter<<<(NCONN + 255) / 256, 256, 0, stream>>>(W, rows, cols, vals, frags);
    swl_gemm<<<GEMM_BLOCKS, 256, 0, stream>>>(x, b, frags, out);
}
